// MarlinInt8Linear_78709570667050
// MI455X (gfx1250) — matrix operand order repaired and checked by reading, not compiled, not run on hardware
//
#include <hip/hip_runtime.h>

#define K_IN 4096
#define N_OUT 16384
#define QGROUP 128
#define NGROUPS (K_IN / QGROUP)

typedef __attribute__((ext_vector_type(16))) __bf16 v16bf;
typedef __attribute__((ext_vector_type(8)))  float  v8f;

struct Frag32B { uint4 lo, hi; };   // 32 bytes == one lane's v16bf fragment

// One-instruction packed fp32x2 -> bf16x2 (RNE), CDNA5 VOP3.
__device__ __forceinline__ unsigned int cvt_pk_bf16(float lo, float hi) {
  unsigned int d;
  asm("v_cvt_pk_bf16_f32 %0, %1, %2" : "=v"(d) : "v"(lo), "v"(hi));
  return d;
}

// Fused dequant+convert+pack via V_FMA_MIXLO_BF16 / V_FMA_MIXHI_BF16
// (all-zero opsel => all three sources read as f32).
__device__ __forceinline__ unsigned int dq_pk_bf16(int qlo, int qhi, float s, float c) {
  float f0 = (float)qlo;   // v_cvt_f32_i32
  float f1 = (float)qhi;   // v_cvt_f32_i32
  unsigned int d;
  asm("v_fma_mixlo_bf16 %0, %1, %2, %3" : "=v"(d) : "v"(f0), "v"(s), "v"(c));
  asm("v_fma_mixhi_bf16 %0, %1, %2, %3" : "+v"(d) : "v"(f1), "v"(s), "v"(c));
  return d;
}

__global__ __launch_bounds__(256, 2)
void marlin_u8b128_bf16_wmma(const float* __restrict__ x,
                             const int*   __restrict__ qw,
                             const float* __restrict__ scales,
                             const float* __restrict__ bias,
                             float* __restrict__ out) {
  // Double-buffered LDS tiles in WMMA-fragment order:
  // [buf][frag16][ksub32][lane][16 bf16]  (2 x 32KB each for A and B = 128KB)
  __shared__ __align__(16) unsigned short lds_a[2][8][4][32][16];
  __shared__ __align__(16) unsigned short lds_b[2][8][4][32][16];
  constexpr int BUFE = 8 * 4 * 32 * 16;   // ushorts per buffer
  constexpr int ROWE = 4 * 32 * 16;       // ushorts per fragment row

  const int tid  = threadIdx.x;
  const int lane = tid & 31;
  const int wv   = tid >> 5;
  const int wm   = wv & 1;    // wave M half   (2)
  const int wn   = wv >> 1;   // wave N column (4)
  // blockIdx.x walks M (scheduled fastest): each 2MB weight tile is reused by
  // 64 consecutive workgroups out of L2; x (134MB) lives in the 192MB L2.
  const int m0 = blockIdx.x * 128;
  const int n0 = blockIdx.y * 128;

  // ---- loop-invariant fill decode: kk = tid&15, row = r*16 + rb, mf == r ----
  const int kk  = tid & 15;            // 8-element run along K (k = kk*8 + j)
  const int rb  = tid >> 4;            // row base 0..15
  const int kf  = kk >> 2, rem = kk & 3;
  const int runA = rem >> 1, hfA = rem & 1;  // A: k = kf*32 + runA*16 + hfA*8 + j
  const int runB = rem >> 1, hfB = rem & 1;  // B: k = kf*32 + runB*16 + hfB*8 + j
  unsigned short* ldsA_w = &lds_a[0][0][kf][rb + 16 * hfA][runA * 8];
  unsigned short* ldsB_w = &lds_b[0][0][kf][rb + 16 * hfB][runB * 8];
  const float* xbase = x  + (size_t)(m0 + rb) * K_IN + kk * 8;
  const int*   qbase = qw + (size_t)(n0 + rb) * K_IN + kk * 8;
  const float* sbase = scales + (size_t)(n0 + rb) * NGROUPS;

  v8f acc[4][2];
#pragma unroll
  for (int i = 0; i < 4; ++i)
#pragma unroll
    for (int j = 0; j < 2; ++j) acc[i][j] = (v8f)0.0f;

  // register staging for the next K-stage (prefetch)
  float4 Ra[8][2];
  int4   Rq[8][2];
  float  Rs[8];

  auto load_stage = [&](int ks) {                       // global -> regs (LOADcnt)
#pragma unroll
    for (int r = 0; r < 8; ++r) {
      const float4* sa = reinterpret_cast<const float4*>(xbase + (size_t)r * 16 * K_IN + ks);
      Ra[r][0] = sa[0];
      Ra[r][1] = sa[1];
      const int4* sq = reinterpret_cast<const int4*>(qbase + (size_t)r * 16 * K_IN + ks);
      Rq[r][0] = sq[0];
      Rq[r][1] = sq[1];
      Rs[r] = sbase[(size_t)r * 16 * NGROUPS + (ks >> 7)];
    }
  };

  auto store_stage = [&](int buf) {                     // regs -> LDS (convert + pack)
#pragma unroll
    for (int r = 0; r < 8; ++r) {
      uint4 p;
      p.x = cvt_pk_bf16(Ra[r][0].x, Ra[r][0].y);
      p.y = cvt_pk_bf16(Ra[r][0].z, Ra[r][0].w);
      p.z = cvt_pk_bf16(Ra[r][1].x, Ra[r][1].y);
      p.w = cvt_pk_bf16(Ra[r][1].z, Ra[r][1].w);
      *(uint4*)(ldsA_w + buf * BUFE + r * ROWE) = p;
      const float s = Rs[r], c = s * -128.0f;
      uint4 w;
      w.x = dq_pk_bf16(Rq[r][0].x, Rq[r][0].y, s, c);
      w.y = dq_pk_bf16(Rq[r][0].z, Rq[r][0].w, s, c);
      w.z = dq_pk_bf16(Rq[r][1].x, Rq[r][1].y, s, c);
      w.w = dq_pk_bf16(Rq[r][1].z, Rq[r][1].w, s, c);
      *(uint4*)(ldsB_w + buf * BUFE + r * ROWE) = w;
    }
  };

  auto compute = [&](int buf) {                          // LDS -> WMMA
#pragma unroll
    for (int kfc = 0; kfc < 4; ++kfc) {
      v16bf a[4], b[2];
#pragma unroll
      for (int i = 0; i < 4; ++i) {
        Frag32B f;
        f.lo = *(const uint4*)&lds_a[buf][wm * 4 + i][kfc][lane][0];
        f.hi = *(const uint4*)&lds_a[buf][wm * 4 + i][kfc][lane][8];
        a[i] = __builtin_bit_cast(v16bf, f);
      }
#pragma unroll
      for (int j = 0; j < 2; ++j) {
        Frag32B f;
        f.lo = *(const uint4*)&lds_b[buf][wn * 2 + j][kfc][lane][0];
        f.hi = *(const uint4*)&lds_b[buf][wn * 2 + j][kfc][lane][8];
        b[j] = __builtin_bit_cast(v16bf, f);
      }
#pragma unroll
      for (int i = 0; i < 4; ++i)
#pragma unroll
        for (int j = 0; j < 2; ++j)
          acc[i][j] = __builtin_amdgcn_wmma_f32_16x16x32_bf16(
              false, a[i], false, b[j], (short)0, acc[i][j], false, false);
    }
  };

  // ---- software pipeline: prefetch(t+1) || compute(t), one barrier/stage ----
  load_stage(0);
  store_stage(0);
  __syncthreads();
#pragma unroll 1
  for (int ks = 0; ks < K_IN - QGROUP; ks += QGROUP) {
    const int buf = (ks >> 7) & 1;
    load_stage(ks + QGROUP);   // global loads in flight during WMMAs
    compute(buf);
    store_stage(buf ^ 1);      // waits LOADcnt here, after the matrix work
    __syncthreads();
  }
  compute(((K_IN >> 7) - 1) & 1);

  // ---- epilogue: bias add + NT store (output is 536MB streamed once; keep L2
  // for the reused x / weight tiles).  C/D layout: N = lane&15, M = r+8*(lane>=16).
  const int lr = lane & 15;
  const int hf = lane >> 4;
#pragma unroll
  for (int j = 0; j < 2; ++j) {
    const int n = n0 + wn * 32 + j * 16 + lr;
    const float bb = bias[n];
#pragma unroll
    for (int i = 0; i < 4; ++i) {
      const int mb = m0 + wm * 64 + i * 16 + hf * 8;
#pragma unroll
      for (int r = 0; r < 8; ++r)
        __builtin_nontemporal_store(acc[i][j][r] + bb,
                                    &out[(size_t)(mb + r) * N_OUT + n]);
    }
  }
}

extern "C" void kernel_launch(void* const* d_in, const int* in_sizes, int n_in,
                              void* d_out, int out_size, void* d_ws, size_t ws_size,
                              hipStream_t stream) {
  const float* x      = (const float*)d_in[0];
  const int*   qw     = (const int*)d_in[1];
  const float* scales = (const float*)d_in[2];
  const float* bias   = (const float*)d_in[3];
  float* out          = (float*)d_out;

  const int M = in_sizes[0] / K_IN;              // 8192
  dim3 grid(M / 128, N_OUT / 128);               // x = M tiles (fastest), y = N tiles
  dim3 block(256);
  marlin_u8b128_bf16_wmma<<<grid, block, 0, stream>>>(x, qw, scales, bias, out);
}
